// VisionEncoderMambaBlock_83408264889054
// MI455X (gfx1250) — matrix operand order repaired and checked by reading, not compiled, not run on hardware
//
#include <hip/hip_runtime.h>
#include <hip/hip_bf16.h>
#include <cstdint>
#include <cstddef>

// Problem constants (match reference)
#define BB  4
#define SS  1024
#define DD  384
#define RR  24
#define NN  16
#define TT  (BB * SS)          // 4096 tokens
#define TD  ((size_t)TT * DD)  // 1,572,864
#define TN  ((size_t)TT * NN)
#define CHN 8                  // scan chunks per sequence
#define CLN (SS / CHN)         // 128 steps per chunk
#define CHF ((size_t)BB * CHN * DD * NN)

// LDS A-panel: 34 rows (32 tokens + 1-row halo each side) x 384 bf16,
// padded to 392 shorts/row (196 dwords = 4 mod 64 banks -> conflict-free
// b128 fragment reads across the 16 M-lanes).
#define AROWS 34
#define APITCH 392

typedef __attribute__((ext_vector_type(16))) __bf16 v16bf;
typedef __attribute__((ext_vector_type(8)))  float  v8f;

union Frag16 { v16bf v; float4 f[2]; };

__device__ __forceinline__ unsigned short f2bf(float x) {
  unsigned u = __float_as_uint(x);
  u += 0x7fffu + ((u >> 16) & 1u);        // round-to-nearest-even
  return (unsigned short)(u >> 16);
}

// ---------------------------------------------------------------------------
// Pack conv weight (O,I,K) f32 -> (K,O,I) bf16 so B-fragment K-values (=i) are
// contiguous per output channel row.
// ---------------------------------------------------------------------------
__global__ void pack_w(const float* __restrict__ src, unsigned short* __restrict__ dst,
                       int taps) {
  int total = taps * DD * DD;
  for (int L = blockIdx.x * blockDim.x + threadIdx.x; L < total;
       L += gridDim.x * blockDim.x) {
    int k = L / (DD * DD);
    int rem = L - k * DD * DD;
    int o = rem / DD, i = rem - o * DD;
    dst[L] = f2bf(src[(o * DD + i) * taps + k]);
  }
}

// ---------------------------------------------------------------------------
// Row LayerNorm over D=384, 128 threads/row (3 elems each), optional bf16 out.
// ---------------------------------------------------------------------------
__global__ void ln_rows(const float* __restrict__ in, const float* __restrict__ g,
                        const float* __restrict__ bt, float* __restrict__ outf,
                        unsigned short* __restrict__ outbf) {
  int row = blockIdx.x, tid = threadIdx.x;
  const float* xr = in + (size_t)row * DD;
  float v0 = xr[tid], v1 = xr[tid + 128], v2 = xr[tid + 256];
  __shared__ float s1[128], s2[128];
  s1[tid] = v0 + v1 + v2;
  s2[tid] = v0 * v0 + v1 * v1 + v2 * v2;
  __syncthreads();
  for (int st = 64; st > 0; st >>= 1) {
    if (tid < st) { s1[tid] += s1[tid + st]; s2[tid] += s2[tid + st]; }
    __syncthreads();
  }
  float mu = s1[0] * (1.0f / DD);
  float var = s2[0] * (1.0f / DD) - mu * mu;
  float rs = rsqrtf(var + 1e-5f);
  float* orf = outf + (size_t)row * DD;
  unsigned short* orb = outbf ? outbf + (size_t)row * DD : nullptr;
#pragma unroll
  for (int e = 0; e < 3; ++e) {
    int d = tid + e * 128;
    float v = (e == 0 ? v0 : (e == 1 ? v1 : v2));
    float y = (v - mu) * rs * g[d] + bt[d];
    orf[d] = y;
    if (orb) orb[d] = f2bf(y);
  }
}

// ---------------------------------------------------------------------------
// Fused conv1d-as-GEMM via WMMA bf16 (f32 accumulate).
//   Y[t,o] = bias[o] + (add?add[t,o]:0) + sum_k sum_i X[t+k-co, i] * W[k][o][i]
// Block: 256 thr = 8 waves. Wave tile 16 tokens x 96 channels (6 accumulators).
// The full halo'd A panel (34 rows x 384 bf16) is brought into LDS ONCE per
// block with gfx1250 async-to-LDS loads (ASYNCcnt), conv taps become LDS row
// offsets, and the wmma main loop runs with no barriers. All six B fragments
// of a K-step are preloaded before the wmma chain so the scheduler can retire
// wmmas against partial load counts instead of s_wait_loadcnt 0.
// ---------------------------------------------------------------------------
__global__ void gemm_conv(const unsigned short* __restrict__ Xbf,
                          const unsigned short* __restrict__ Wbf,
                          const float* __restrict__ bias, int taps,
                          float* __restrict__ outf,
                          unsigned short* __restrict__ outbf,
                          const float* __restrict__ addf) {
  __shared__ __align__(16) unsigned short ldsA[AROWS * APITCH];
  const int tid = threadIdx.x;
  const int lane = tid & 31, w = tid >> 5;
  const int mt = w & 1, og = w >> 1;          // token sub-tile, out-channel group
  const int nlane = lane & 15, hi = lane >> 4;
  const int t0 = blockIdx.x * 32;
  const int b = t0 / SS, s0 = t0 - b * SS;    // block fully inside one batch
  const int co = (taps - 1) >> 1;

  // ---- one-shot async panel load: rows r=0..33 hold tokens s0-1 .. s0+32 ----
  for (int idx = tid; idx < AROWS * 48; idx += 256) {
    int r = idx / 48, ch = idx - r * 48;      // 48 x b128 chunks per 384-ch row
    int s = s0 + r - 1;
    unsigned short* lp = ldsA + r * APITCH + ch * 8;
    if ((unsigned)s < (unsigned)SS) {
      const unsigned short* gp = Xbf + ((size_t)(b * SS + s) * DD + ch * 8);
      unsigned ldsoff = (unsigned)(uintptr_t)lp;
      asm volatile("global_load_async_to_lds_b128 %0, %1, off"
                   :: "v"(ldsoff), "v"((unsigned long long)(uintptr_t)gp)
                   : "memory");
    } else {
      uint4 z; z.x = 0u; z.y = 0u; z.z = 0u; z.w = 0u;  // conv zero padding
      *(uint4*)lp = z;
    }
  }
  asm volatile("s_wait_asynccnt 0x0" ::: "memory");
  __syncthreads();

  v8f zero = {0.f, 0.f, 0.f, 0.f, 0.f, 0.f, 0.f, 0.f};
  v8f acc[6];
#pragma unroll
  for (int c = 0; c < 6; ++c) acc[c] = zero;

  // row in LDS for this wave's A fragment at tap k: (m + k + 1 - co)
  const int rbase = mt * 16 + nlane + 1 - co;

  for (int k = 0; k < taps; ++k) {
    const unsigned short* ar = ldsA + (rbase + k) * APITCH;
    const unsigned short* wrow = Wbf + ((size_t)k * DD + og * 96 + nlane) * DD + hi * 16;
    for (int i0 = 0; i0 < DD; i0 += 32) {
      // A fragment (16x32 bf16): lane-half selects K 0-15 vs 16-31
      Frag16 afr;
      afr.f[0] = *(const float4*)(ar + i0 + hi * 16);
      afr.f[1] = *(const float4*)(ar + i0 + hi * 16 + 8);
      // preload all six B fragments (independent of the wmma chain)
      Frag16 bfr[6];
#pragma unroll
      for (int c = 0; c < 6; ++c) {
        const unsigned short* wp = wrow + (size_t)(c * 16) * DD + i0;
        bfr[c].f[0] = *(const float4*)(wp);
        bfr[c].f[1] = *(const float4*)(wp + 8);
      }
#pragma unroll
      for (int c = 0; c < 6; ++c)
        acc[c] = __builtin_amdgcn_wmma_f32_16x16x32_bf16(
            false, afr.v, false, bfr[c].v, (short)0, acc[c], false, false);
    }
  }

  const bool wf = (outf != nullptr), wb = (outbf != nullptr), wa = (addf != nullptr);
#pragma unroll
  for (int c = 0; c < 6; ++c) {
#pragma unroll
    for (int r = 0; r < 8; ++r) {
      int m = r + hi * 8;                      // C/D layout: M = vgpr + 8*laneHi
      size_t idx = (size_t)(t0 + mt * 16 + m) * DD + (og * 96 + c * 16 + nlane);
      float v = acc[c][r] + bias[og * 96 + c * 16 + nlane];
      if (wa) v += addf[idx];
      if (wf) outf[idx] = v;
      if (wb) outbf[idx] = f2bf(v);
    }
  }
}

// ---------------------------------------------------------------------------
// Per-token SSM projections: dbc = h @ w_dbc^T ; delta = softplus(dr@w_dt^T+b)
// Block = 384 threads, one token per block (0.5 GF total -> VALU is fine).
// ---------------------------------------------------------------------------
__global__ void ssm_prep(const float* __restrict__ hln, const float* __restrict__ w_dbc,
                         const float* __restrict__ w_dt, const float* __restrict__ b_dt,
                         float* __restrict__ delta, float* __restrict__ Bm,
                         float* __restrict__ Cm) {
  int t = blockIdx.x, tid = threadIdx.x;
  __shared__ float sdbc[RR + 2 * NN];
  const float* xr = hln + (size_t)t * DD;
  if (tid < RR + 2 * NN) {
    const float* wr = w_dbc + (size_t)tid * DD;
    float s = 0.f;
    for (int i = 0; i < DD; ++i) s += wr[i] * xr[i];
    sdbc[tid] = s;
  }
  __syncthreads();
  {
    float a = b_dt[tid];
    const float* wd = w_dt + (size_t)tid * RR;
#pragma unroll
    for (int r = 0; r < RR; ++r) a += sdbc[r] * wd[r];
    float sp = (a > 20.f) ? a : log1pf(__expf(a));   // softplus
    delta[(size_t)t * DD + tid] = sp;
  }
  if (tid < NN) Bm[(size_t)t * NN + tid] = sdbc[RR + tid];
  else if (tid < 2 * NN) Cm[(size_t)t * NN + (tid - NN)] = sdbc[RR + NN + (tid - NN)];
}

// ---------------------------------------------------------------------------
// Chunked associative scan. Phase 1: per-chunk (prodA, h). Lane = one d.
// flip!=0 reads the sequence reversed (backward direction, no un-flip, per ref).
// ---------------------------------------------------------------------------
__global__ void scan_part1(const float* __restrict__ delta, const float* __restrict__ hln,
                           const float* __restrict__ Bm, const float* __restrict__ A_log,
                           float* __restrict__ chA, float* __restrict__ chH, int flip) {
  int d = threadIdx.x, b = blockIdx.x, c = blockIdx.y;
  float a[NN], pA[NN], h[NN];
#pragma unroll
  for (int n = 0; n < NN; ++n) {
    a[n] = -__expf(A_log[d * NN + n]);
    pA[n] = 1.f; h[n] = 0.f;
  }
  for (int sl = 0; sl < CLN; ++sl) {
    int t = c * CLN + sl;
    int sx = flip ? (SS - 1 - t) : t;
    size_t id = ((size_t)b * SS + sx) * DD + d;
    float dl = delta[id];
    float dx = dl * hln[id];
    const float* Bp = Bm + ((size_t)b * SS + sx) * NN;
#pragma unroll
    for (int n = 0; n < NN; ++n) {
      float e = __expf(dl * a[n]);
      h[n] = e * h[n] + dx * Bp[n];
      pA[n] *= e;
    }
  }
  size_t o = (((size_t)b * CHN + c) * DD + d) * NN;
#pragma unroll
  for (int n = 0; n < NN; ++n) { chA[o + n] = pA[n]; chH[o + n] = h[n]; }
}

// Phase 2: tiny serial prefix across the 8 chunks -> carry-in h per chunk.
__global__ void scan_part2(const float* __restrict__ chA, const float* __restrict__ chH,
                           float* __restrict__ carry) {
  int id = blockIdx.x * blockDim.x + threadIdx.x;
  if (id >= BB * DD * NN) return;
  int n = id & (NN - 1);
  int d = (id >> 4) % DD;
  int b = id / (DD * NN);
  float h = 0.f;
  for (int c = 0; c < CHN; ++c) {
    size_t o = (((size_t)b * CHN + c) * DD + d) * NN + n;
    carry[o] = h;
    h = chA[o] * h + chH[o];
  }
}

// Phase 3: rerun local scan with carry-in, emit y = <h,C> + D*x at index t.
__global__ void scan_part3(const float* __restrict__ delta, const float* __restrict__ hln,
                           const float* __restrict__ Bm, const float* __restrict__ Cm,
                           const float* __restrict__ A_log, const float* __restrict__ Dv,
                           const float* __restrict__ carry, float* __restrict__ y,
                           int flip) {
  int d = threadIdx.x, b = blockIdx.x, c = blockIdx.y;
  float a[NN], h[NN];
  size_t o = (((size_t)b * CHN + c) * DD + d) * NN;
#pragma unroll
  for (int n = 0; n < NN; ++n) {
    a[n] = -__expf(A_log[d * NN + n]);
    h[n] = carry[o + n];
  }
  float Dd = Dv[d];
  for (int sl = 0; sl < CLN; ++sl) {
    int t = c * CLN + sl;
    int sx = flip ? (SS - 1 - t) : t;
    size_t id = ((size_t)b * SS + sx) * DD + d;
    float dl = delta[id];
    float xv = hln[id];
    float dx = dl * xv;
    const float* Bp = Bm + ((size_t)b * SS + sx) * NN;
    const float* Cp = Cm + ((size_t)b * SS + sx) * NN;
    float accv = 0.f;
#pragma unroll
    for (int n = 0; n < NN; ++n) {
      float e = __expf(dl * a[n]);
      h[n] = e * h[n] + dx * Bp[n];
      accv += h[n] * Cp[n];
    }
    y[((size_t)b * SS + t) * DD + d] = accv + Dd * xv;
  }
}

// (y1+y2) * silu(z1) -> bf16 for the final 1x1 conv
__global__ void combine_silu(const float* __restrict__ y1, const float* __restrict__ y2,
                             const float* __restrict__ z1, unsigned short* __restrict__ x3bf) {
  size_t i = (size_t)blockIdx.x * blockDim.x + threadIdx.x;
  if (i >= TD) return;
  float z = z1[i];
  float s = z / (1.f + __expf(-z));
  x3bf[i] = f2bf((y1[i] + y2[i]) * s);
}

// ---------------------------------------------------------------------------
extern "C" void kernel_launch(void* const* d_in, const int* in_sizes, int n_in,
                              void* d_out, int out_size, void* d_ws, size_t ws_size,
                              hipStream_t stream) {
  (void)in_sizes; (void)n_in; (void)out_size; (void)ws_size;
  const float* x      = (const float*)d_in[0];
  const float* w_p1   = (const float*)d_in[1];  const float* b_p1 = (const float*)d_in[2];
  const float* w_p2   = (const float*)d_in[3];  const float* b_p2 = (const float*)d_in[4];
  const float* w_p3   = (const float*)d_in[5];  const float* b_p3 = (const float*)d_in[6];
  const float* w_cf   = (const float*)d_in[7];  const float* b_cf = (const float*)d_in[8];
  const float* w_cb   = (const float*)d_in[9];  const float* b_cb = (const float*)d_in[10];
  const float* g_n    = (const float*)d_in[11]; const float* bt_n = (const float*)d_in[12];
  const float* g_nf   = (const float*)d_in[13]; const float* bt_nf= (const float*)d_in[14];
  const float* g_nb   = (const float*)d_in[15]; const float* bt_nb= (const float*)d_in[16];
  const float* w_dbc1 = (const float*)d_in[17]; const float* w_dt1= (const float*)d_in[18];
  const float* b_dt1  = (const float*)d_in[19]; const float* A_l1 = (const float*)d_in[20];
  const float* D1     = (const float*)d_in[21];
  const float* w_dbc2 = (const float*)d_in[22]; const float* w_dt2= (const float*)d_in[23];
  const float* b_dt2  = (const float*)d_in[24]; const float* A_l2 = (const float*)d_in[25];
  const float* D2     = (const float*)d_in[26];

  char* base = (char*)d_ws;
  size_t off = 0;
  auto take = [&](size_t bytes) -> char* {
    char* r = base + off;
    off += (bytes + 255) & ~(size_t)255;
    return r;
  };
  float* xn  = (float*)take(4 * TD);   // layernormed input (reused as y1 later)
  float* z1  = (float*)take(4 * TD);   // gate conv output (pre-silu)
  float* tf  = (float*)take(4 * TD);   // cf conv out, then reused as delta_f
  float* tb  = (float*)take(4 * TD);   // cb conv out, then reused as delta_b
  float* hf  = (float*)take(4 * TD);   // LN(cf conv)
  float* hb  = (float*)take(4 * TD);   // LN(cb conv)
  float* Bfm = (float*)take(4 * TN);
  float* Cfm = (float*)take(4 * TN);
  float* Bbm = (float*)take(4 * TN);
  float* Cbm = (float*)take(4 * TN);
  float* chAf = (float*)take(4 * CHF);
  float* chHf = (float*)take(4 * CHF);
  float* caf  = (float*)take(4 * CHF);
  float* chAb = (float*)take(4 * CHF);
  float* chHb = (float*)take(4 * CHF);
  float* cab  = (float*)take(4 * CHF);
  unsigned short* xnbf = (unsigned short*)take(2 * TD);  // reused as x3_bf16
  unsigned short* xpbf = (unsigned short*)take(2 * TD);
  unsigned short* Wp1 = (unsigned short*)take(2 * (size_t)DD * DD * 3);
  unsigned short* Wp2 = (unsigned short*)take(2 * (size_t)DD * DD * 3);
  unsigned short* Wcf = (unsigned short*)take(2 * (size_t)DD * DD);
  unsigned short* Wcb = (unsigned short*)take(2 * (size_t)DD * DD);
  unsigned short* Wp3 = (unsigned short*)take(2 * (size_t)DD * DD);
  float* delta_f = tf;
  float* delta_b = tb;
  float* y1 = xn;
  float* y2 = (float*)d_out;           // scratch until final conv overwrites it
  unsigned short* x3bf = xnbf;

  // 1) weight packing to bf16 [k][o][i]
  pack_w<<<512, 256, 0, stream>>>(w_p1, Wp1, 3);
  pack_w<<<512, 256, 0, stream>>>(w_p2, Wp2, 3);
  pack_w<<<256, 256, 0, stream>>>(w_cf, Wcf, 1);
  pack_w<<<256, 256, 0, stream>>>(w_cb, Wcb, 1);
  pack_w<<<256, 256, 0, stream>>>(w_p3, Wp3, 1);

  // 2) LN(x) -> f32 + bf16
  ln_rows<<<TT, 128, 0, stream>>>(x, g_n, bt_n, xn, xnbf);

  // 3) K=3 convs as WMMA GEMMs: z1 (gate) and xp (to bf16 only)
  gemm_conv<<<TT / 32, 256, 0, stream>>>(xnbf, Wp1, b_p1, 3, z1, nullptr, nullptr);
  gemm_conv<<<TT / 32, 256, 0, stream>>>(xnbf, Wp2, b_p2, 3, nullptr, xpbf, nullptr);

  // 4) K=1 convs for the two directions (pointwise, so flip commutes; scan flips reads)
  gemm_conv<<<TT / 32, 256, 0, stream>>>(xpbf, Wcf, b_cf, 1, tf, nullptr, nullptr);
  gemm_conv<<<TT / 32, 256, 0, stream>>>(xpbf, Wcb, b_cb, 1, tb, nullptr, nullptr);

  // 5) per-direction LN
  ln_rows<<<TT, 128, 0, stream>>>(tf, g_nf, bt_nf, hf, nullptr);
  ln_rows<<<TT, 128, 0, stream>>>(tb, g_nb, bt_nb, hb, nullptr);

  // 6) SSM projections (delta overwrites tf/tb, which are now dead)
  ssm_prep<<<TT, DD, 0, stream>>>(hf, w_dbc1, w_dt1, b_dt1, delta_f, Bfm, Cfm);
  ssm_prep<<<TT, DD, 0, stream>>>(hb, w_dbc2, w_dt2, b_dt2, delta_b, Bbm, Cbm);

  // 7) chunked associative scan (forward + flipped backward)
  dim3 sg(BB, CHN);
  scan_part1<<<sg, DD, 0, stream>>>(delta_f, hf, Bfm, A_l1, chAf, chHf, 0);
  scan_part1<<<sg, DD, 0, stream>>>(delta_b, hb, Bbm, A_l2, chAb, chHb, 1);
  int p2n = (BB * DD * NN + 255) / 256;
  scan_part2<<<p2n, 256, 0, stream>>>(chAf, chHf, caf);
  scan_part2<<<p2n, 256, 0, stream>>>(chAb, chHb, cab);
  scan_part3<<<sg, DD, 0, stream>>>(delta_f, hf, Bfm, Cfm, A_l1, D1, caf, y1, 0);
  scan_part3<<<sg, DD, 0, stream>>>(delta_b, hb, Bbm, Cbm, A_l2, D2, cab, y2, 1);

  // 8) (y1+y2)*silu(z1) -> bf16
  combine_silu<<<(int)((TD + 255) / 256), 256, 0, stream>>>(y1, y2, z1, x3bf);

  // 9) final 1x1 conv + bias + skip -> d_out (f32)
  gemm_conv<<<TT / 32, 256, 0, stream>>>(x3bf, Wp3, b_p3, 1, (float*)d_out, nullptr, x);
}
